// KnapsackGAT_58875411694098
// MI455X (gfx1250) — matrix operand order repaired and checked by reading, not compiled, not run on hardware
//
#include <hip/hip_runtime.h>
#include <hip/hip_bf16.h>
#include <math.h>

#define N_NODES 100000
#define N_EDGES 800000
#define HID 128
#define GROWS 32            // rows per GEMM block (100000 % 32 == 0)

typedef __attribute__((ext_vector_type(16))) _Float16 v16h;
typedef __attribute__((ext_vector_type(8)))  float    v8f;

// ---------- small helpers ----------
__global__ void k_cvt_f16(_Float16* __restrict__ dst, const float* __restrict__ src, int n) {
    int i = blockIdx.x * blockDim.x + threadIdx.x;
    if (i < n) dst[i] = (_Float16)src[i];
}

__global__ void k_fill(float* __restrict__ p, int n, float v) {
    int i = blockIdx.x * blockDim.x + threadIdx.x;
    if (i < n) p[i] = v;
}

// ---------- encoder layer 1 (both type branches) ----------
__global__ void k_enc1(float* __restrict__ item_h1, float* __restrict__ con_h1,
                       const float* __restrict__ x,
                       const float* __restrict__ ew1, const float* __restrict__ eb1,
                       const float* __restrict__ cw1, const float* __restrict__ cb1) {
    int idx = blockIdx.x * blockDim.x + threadIdx.x;
    if (idx >= N_NODES * HID) return;
    int n = idx / HID, j = idx % HID;
    float x0 = x[n * 2 + 0], x1 = x[n * 2 + 1];
    float vi = ew1[j * 2 + 0] * x0 + ew1[j * 2 + 1] * x1 + eb1[j];
    float vc = cw1[j] * x0 + cb1[j];
    item_h1[idx] = fmaxf(vi, 0.f);
    con_h1[idx]  = fmaxf(vc, 0.f);
}

// h (in-place over con_h) = select by node type
__global__ void k_select(float* __restrict__ h, const float* __restrict__ item_h,
                         const int* __restrict__ types) {
    int idx = blockIdx.x * blockDim.x + threadIdx.x;
    if (idx >= N_NODES * HID) return;
    int t = types[idx / HID];
    h[idx] = (t == 0) ? item_h[idx] : ((t == 1) ? h[idx] : 0.f);
}

// ---------- WMMA GEMM: out[M,outCols] = act(in[M,128] @ W^T + bias) ----------
// W stored as f16 row-major [outCols,128], so B[k][n] = W[n][k].
// Block: 256 threads, 32 rows (two 16-row tiles); wave w owns cols [16w,16w+16).
// A tile staged raw-f32 into LDS via GLOBAL_LOAD_ASYNC_TO_LDS_B128 (ASYNCcnt),
// converted to f16 during fragment build (co-executes with WMMA).
__global__ void __launch_bounds__(256)
k_gemm(float* __restrict__ out, const float* __restrict__ in,
       const _Float16* __restrict__ w16, const float* __restrict__ bias,
       int outCols, int doRelu) {
    __shared__ float Af[GROWS][HID];            // 32x128 f32 tile, 16KB
    const int rowBase = blockIdx.x * GROWS;
    const int tid = threadIdx.x;

    // stage: 32*128 f32 = 1024 x 16B; 256 threads -> 4 x b128 each
#if defined(__gfx1250__)
    {
        #pragma unroll
        for (int pass = 0; pass < 4; ++pass) {
            int v4 = pass * 256 + tid;           // float4 index 0..1023
            int r = v4 >> 5, c = (v4 & 31) * 4;  // 32 float4 per row
            unsigned lds = (unsigned)(size_t)&Af[r][c];          // low 32b = DS addr
            unsigned long long ga =
                (unsigned long long)(in + (size_t)(rowBase + r) * HID + c);
            asm volatile("global_load_async_to_lds_b128 %0, %1, off"
                         :: "v"(lds), "v"(ga) : "memory");
        }
        asm volatile("s_wait_asynccnt 0x0" ::: "memory");
    }
#else
    for (int pass = 0; pass < 4; ++pass) {
        int v4 = pass * 256 + tid;
        int r = v4 >> 5, c = (v4 & 31) * 4;
        *(float4*)&Af[r][c] = *(const float4*)(in + (size_t)(rowBase + r) * HID + c);
    }
#endif
    __syncthreads();

    const int wave = tid >> 5, lane = tid & 31;
    const int colBase = wave * 16;
    if (colBase >= outCols) return;              // wave-uniform (EXEC stays all-1s)
    const int nn = lane & 15;                    // A row m / B col n
    const int hi = lane >> 4;

    v8f acc0 = {}, acc1 = {};
    #pragma unroll
    for (int kk = 0; kk < 4; ++kk) {
        // B fragment: k = e + hi*16 (+kk*32); lane owns W row (colBase+nn)
        const _Float16* brow = w16 + (size_t)(colBase + nn) * HID + kk * 32 + hi * 16;
        v16h b = *(const v16h*)brow;             // 32B-aligned global load
        #pragma unroll
        for (int t = 0; t < 2; ++t) {            // two 16-row tiles share b
            // A fragment: k = e + hi*16 (+kk*32)
            const float* arow = &Af[t * 16 + nn][kk * 32 + hi * 16];
            float4 p0 = *(const float4*)(arow);
            float4 p1 = *(const float4*)(arow + 4);
            float4 p2 = *(const float4*)(arow + 8);
            float4 p3 = *(const float4*)(arow + 12);
            v16h a;
            a[0]  = (_Float16)p0.x; a[1]  = (_Float16)p0.y;
            a[2]  = (_Float16)p0.z; a[3]  = (_Float16)p0.w;
            a[4]  = (_Float16)p1.x; a[5]  = (_Float16)p1.y;
            a[6]  = (_Float16)p1.z; a[7]  = (_Float16)p1.w;
            a[8]  = (_Float16)p2.x; a[9]  = (_Float16)p2.y;
            a[10] = (_Float16)p2.z; a[11] = (_Float16)p2.w;
            a[12] = (_Float16)p3.x; a[13] = (_Float16)p3.y;
            a[14] = (_Float16)p3.z; a[15] = (_Float16)p3.w;
            if (t == 0)
                acc0 = __builtin_amdgcn_wmma_f32_16x16x32_f16(
                    false, a, false, b, (short)0, acc0, false, false);
            else
                acc1 = __builtin_amdgcn_wmma_f32_16x16x32_f16(
                    false, a, false, b, (short)0, acc1, false, false);
        }
    }

    const int col = colBase + nn;
    const float bv = bias ? bias[col] : 0.f;
    #pragma unroll
    for (int r = 0; r < 8; ++r) {                // C/D: VGPR r -> row r + hi*8
        int m = r + hi * 8;
        float v0 = acc0[r] + bv;
        float v1 = acc1[r] + bv;
        if (doRelu) { v0 = fmaxf(v0, 0.f); v1 = fmaxf(v1, 0.f); }
        out[(size_t)(rowBase + m) * outCols + col]      = v0;
        out[(size_t)(rowBase + 16 + m) * outCols + col] = v1;
    }
}

// ---------- attention coefficients: a_s/a_d[n,h] = <hl[n,h,:], att[h,:]> ----------
__global__ void k_attcoef(float* __restrict__ a_s, float* __restrict__ a_d,
                          const float* __restrict__ hl,
                          const float* __restrict__ att_s, const float* __restrict__ att_d,
                          int H, int C) {
    int idx = blockIdx.x * blockDim.x + threadIdx.x;
    if (idx >= N_NODES * H) return;
    int n = idx / H, h = idx % H;
    const float* row = hl + (size_t)n * HID + h * C;
    const float* ws = att_s + h * C;
    const float* wd = att_d + h * C;
    float ss = 0.f, sd = 0.f;
    for (int c = 0; c < C; ++c) { float v = row[c]; ss += v * ws[c]; sd += v * wd[c]; }
    a_s[idx] = ss; a_d[idx] = sd;
}

// ---------- edge phases ----------
__device__ __forceinline__ float lrelu(float v) { return v > 0.f ? v : 0.2f * v; }

__device__ __forceinline__ void atomicMaxF(float* addr, float v) {
    if (v >= 0.f) atomicMax((int*)addr, __float_as_int(v));
    else          atomicMin((unsigned int*)addr, __float_as_uint(v));
}

__device__ __forceinline__ void edge_endpoints(const int* ei, long long e, int& s, int& d) {
    if (e < N_EDGES) { s = ei[e]; d = ei[N_EDGES + e]; }
    else             { s = d = (int)(e - N_EDGES); }   // self loops appended
}

__global__ void k_edge_max(const int* __restrict__ ei, const float* __restrict__ a_s,
                           const float* __restrict__ a_d, float* __restrict__ amax, int H) {
    long long t = (long long)blockIdx.x * blockDim.x + threadIdx.x;
    long long total = (long long)(N_EDGES + N_NODES) * H;
    if (t >= total) return;
    int h = (int)(t % H); long long e = t / H;
    int s, d; edge_endpoints(ei, e, s, d);
    float alpha = lrelu(a_s[s * H + h] + a_d[d * H + h]);
    atomicMaxF(&amax[d * H + h], alpha);
}

__global__ void k_edge_sum(const int* __restrict__ ei, const float* __restrict__ a_s,
                           const float* __restrict__ a_d, const float* __restrict__ amax,
                           float* __restrict__ denom, int H) {
    long long t = (long long)blockIdx.x * blockDim.x + threadIdx.x;
    long long total = (long long)(N_EDGES + N_NODES) * H;
    if (t >= total) return;
    int h = (int)(t % H); long long e = t / H;
    int s, d; edge_endpoints(ei, e, s, d);
    float alpha = lrelu(a_s[s * H + h] + a_d[d * H + h]);
    atomicAdd(&denom[d * H + h], __expf(alpha - amax[d * H + h]));
}

__global__ void k_edge_msg(const int* __restrict__ ei, const float* __restrict__ a_s,
                           const float* __restrict__ a_d, const float* __restrict__ amax,
                           const float* __restrict__ denom, const float* __restrict__ hl,
                           float* __restrict__ accum, int H, int C) {
    const int CH = C / 32;                        // 32-channel chunks per thread
    long long t = (long long)blockIdx.x * blockDim.x + threadIdx.x;
    long long total = (long long)(N_EDGES + N_NODES) * H * CH;
    if (t >= total) return;
    int q = (int)(t % CH);
    int h = (int)((t / CH) % H);
    long long e = t / ((long long)CH * H);
    int s, d; edge_endpoints(ei, e, s, d);
    float alpha = lrelu(a_s[s * H + h] + a_d[d * H + h]);
    float att = __expf(alpha - amax[d * H + h]) / denom[d * H + h];
    const float* hs = hl + (size_t)s * HID + h * C + q * 32;
    float* od = accum + (size_t)d * HID + h * C + q * 32;
    #pragma unroll
    for (int c = 0; c < 32; ++c) atomicAdd(od + c, hs[c] * att);
}

// h = (residual ? h : 0) + relu(accum + bias)
__global__ void k_finish(float* __restrict__ h, const float* __restrict__ accum,
                         const float* __restrict__ bias, int residual) {
    int idx = blockIdx.x * blockDim.x + threadIdx.x;
    if (idx >= N_NODES * HID) return;
    float v = fmaxf(accum[idx] + bias[idx % HID], 0.f);
    h[idx] = (residual ? h[idx] : 0.f) + v;
}

// ---------- decoder head: sigmoid(d2 @ dw3^T + db3) * (type==0) ----------
__global__ void k_final(float* __restrict__ out, const float* __restrict__ d2,
                        const float* __restrict__ w3, const float* __restrict__ b3,
                        const int* __restrict__ types) {
    int n = blockIdx.x * blockDim.x + threadIdx.x;
    if (n >= N_NODES) return;
    const float* row = d2 + (size_t)n * 64;
    float z = b3[0];
    for (int c = 0; c < 64; ++c) z += row[c] * w3[c];
    float p = 1.f / (1.f + __expf(-z));
    out[n] = (types[n] == 0) ? p : 0.f;
}

extern "C" void kernel_launch(void* const* d_in, const int* in_sizes, int n_in,
                              void* d_out, int out_size, void* d_ws, size_t ws_size,
                              hipStream_t stream) {
    const float* x    = (const float*)d_in[0];
    const int*   typ  = (const int*)d_in[1];
    const int*   ei   = (const int*)d_in[2];
    const float* ew1  = (const float*)d_in[3];  const float* eb1 = (const float*)d_in[4];
    const float* ew2  = (const float*)d_in[5];  const float* eb2 = (const float*)d_in[6];
    const float* cw1  = (const float*)d_in[7];  const float* cb1 = (const float*)d_in[8];
    const float* cw2  = (const float*)d_in[9];  const float* cb2 = (const float*)d_in[10];
    const float* lin[3] = {(const float*)d_in[11], (const float*)d_in[15], (const float*)d_in[19]};
    const float* as_[3] = {(const float*)d_in[12], (const float*)d_in[16], (const float*)d_in[20]};
    const float* ad_[3] = {(const float*)d_in[13], (const float*)d_in[17], (const float*)d_in[21]};
    const float* bb[3]  = {(const float*)d_in[14], (const float*)d_in[18], (const float*)d_in[22]};
    const float* dw1 = (const float*)d_in[23]; const float* db1 = (const float*)d_in[24];
    const float* dw2 = (const float*)d_in[25]; const float* db2 = (const float*)d_in[26];
    const float* dw3 = (const float*)d_in[27]; const float* db3 = (const float*)d_in[28];

    // workspace bump allocator (256B aligned)
    char* p = (char*)d_ws;
    auto alloc = [&](size_t bytes) -> char* {
        char* r = p; p += (bytes + 255) & ~(size_t)255; return r;
    };
    const float* Wsrc[7] = {ew2, cw2, lin[0], lin[1], lin[2], dw1, dw2};
    int Wcnt[7] = {HID*HID, HID*HID, HID*HID, HID*HID, HID*HID, HID*HID, 64*HID};
    _Float16* Wh[7];
    for (int i = 0; i < 7; ++i) Wh[i] = (_Float16*)alloc((size_t)Wcnt[i] * sizeof(_Float16));
    float* buf0  = (float*)alloc((size_t)N_NODES * HID * 4);
    float* buf1  = (float*)alloc((size_t)N_NODES * HID * 4);
    float* buf2  = (float*)alloc((size_t)N_NODES * HID * 4);
    float* a_s   = (float*)alloc((size_t)N_NODES * 4 * 4);
    float* a_d   = (float*)alloc((size_t)N_NODES * 4 * 4);
    float* amax  = (float*)alloc((size_t)N_NODES * 4 * 4);
    float* denom = (float*)alloc((size_t)N_NODES * 4 * 4);

    const int TB = 256;
    auto nb = [&](long long n) { return (unsigned)((n + TB - 1) / TB); };
    const unsigned GB = N_NODES / GROWS;   // 3125 GEMM blocks

    // f16 weight copies (L2-resident, reused every GEMM tile)
    for (int i = 0; i < 7; ++i)
        k_cvt_f16<<<nb(Wcnt[i]), TB, 0, stream>>>(Wh[i], Wsrc[i], Wcnt[i]);

    // heterogeneous encoder
    k_enc1<<<nb((long long)N_NODES * HID), TB, 0, stream>>>(buf0, buf1, x, ew1, eb1, cw1, cb1);
    k_gemm<<<GB, 256, 0, stream>>>(buf2, buf0, Wh[0], eb2, HID, 1);  // item_h
    k_gemm<<<GB, 256, 0, stream>>>(buf0, buf1, Wh[1], cb2, HID, 1);  // con_h
    k_select<<<nb((long long)N_NODES * HID), TB, 0, stream>>>(buf0, buf2, typ);

    // three GAT layers
    const int Hs[3] = {4, 4, 1}, Cs[3] = {32, 32, 128};
    for (int l = 0; l < 3; ++l) {
        const int H = Hs[l], C = Cs[l];
        k_gemm<<<GB, 256, 0, stream>>>(buf1, buf0, Wh[2 + l], nullptr, HID, 0); // hl
        k_attcoef<<<nb((long long)N_NODES * H), TB, 0, stream>>>(a_s, a_d, buf1, as_[l], ad_[l], H, C);
        k_fill<<<nb(N_NODES * H), TB, 0, stream>>>(amax, N_NODES * H, -INFINITY);
        k_fill<<<nb(N_NODES * H), TB, 0, stream>>>(denom, N_NODES * H, 0.f);
        k_fill<<<nb((long long)N_NODES * HID), TB, 0, stream>>>(buf2, N_NODES * HID, 0.f);
        const long long EH = (long long)(N_EDGES + N_NODES) * H;
        k_edge_max<<<nb(EH), TB, 0, stream>>>(ei, a_s, a_d, amax, H);
        k_edge_sum<<<nb(EH), TB, 0, stream>>>(ei, a_s, a_d, amax, denom, H);
        k_edge_msg<<<nb(EH * (C / 32)), TB, 0, stream>>>(ei, a_s, a_d, amax, denom, buf1, buf2, H, C);
        k_finish<<<nb((long long)N_NODES * HID), TB, 0, stream>>>(buf0, buf2, bb[l], l > 0 ? 1 : 0);
    }

    // decoder
    k_gemm<<<GB, 256, 0, stream>>>(buf1, buf0, Wh[5], db1, HID, 1);
    k_gemm<<<GB, 256, 0, stream>>>(buf2, buf1, Wh[6], db2, 64, 1);
    k_final<<<nb(N_NODES), TB, 0, stream>>>((float*)d_out, buf2, dw3, db3, typ);
}
